// Block_29695403885071
// MI455X (gfx1250) — matrix operand order repaired and checked by reading, not compiled, not run on hardware
//
#include <hip/hip_runtime.h>
#include <hip/hip_bf16.h>
#include <math.h>
#include <stdint.h>

typedef __bf16 bf16_t;
typedef bf16_t v16bf __attribute__((ext_vector_type(16)));
typedef float  v8f   __attribute__((ext_vector_type(8)));
typedef unsigned int u32x4 __attribute__((ext_vector_type(4)));
typedef int          i32x8 __attribute__((ext_vector_type(8)));
typedef int          i32x4 __attribute__((ext_vector_type(4)));

union Frag16 { v16bf v; unsigned int u[8]; };
union CvtU   { unsigned int u; bf16_t h[2]; };

#define WMMA_BF16(a,b,c) \
  __builtin_amdgcn_wmma_f32_16x16x32_bf16(false,(a),false,(b),(short)0,(c),false,false)

#if __has_builtin(__builtin_amdgcn_tensor_load_to_lds) && \
    __has_builtin(__builtin_amdgcn_s_wait_tensorcnt)
#define USE_TDM 1
#else
#define USE_TDM 0
#endif

#if USE_TDM
// ---------------------------------------------------------------------------
// Issue one TDM 2-D tile load: 128 rows x 32 bf16, row stride K elements,
// destination rows padded 64B data + 16B pad (matches 40-bf16 LDS rows).
// D# layout per CDNA5 ISA ch.8 (group0: count/lds/global/type, group1: dims).
// This toolchain exposes the 6-arg builtin (g0, g1, g2, g3, g4, cpol);
// groups 2-4 are unused for a 2-D tensor -> zero-filled.
// ---------------------------------------------------------------------------
__device__ __forceinline__ void tdm_load_tile(const bf16_t* gptr,
                                              unsigned lds_off, int K)
{
  unsigned long long ga = (unsigned long long)(uintptr_t)gptr;
  u32x4 g0;
  g0[0] = 1u;                                   // count=1, user descriptor
  g0[1] = lds_off;                              // lds_addr (bytes)
  g0[2] = (unsigned)(ga & 0xFFFFFFFFu);         // global_addr[31:0]
  g0[3] = (unsigned)((ga >> 32) & 0x01FFFFFFu)  // global_addr[56:32]
        | 0x80000000u;                          // type=2 ("image")
  i32x8 g1;
  g1[0] = (int)((1u << 16)      // data_size = 2 bytes
              | (1u << 20)      // pad_enable
              | (3u << 22)      // pad_interval: 16 dwords (=64B of data)
              | (3u << 25));    // pad_amount:   4 dwords (=16B pad)
  g1[1] = (int)(32u  << 16);    // tensor_dim0[15:0]  = 32   (bits 79:48)
  g1[2] = (int)(128u << 16);    // tensor_dim1[15:0]  = 128  (bits 111:80)
  g1[3] = (int)(32u  << 16);    // tile_dim0 = 32            (bits 127:112)
  g1[4] = 128;                  // tile_dim1 = 128           (bits 143:128)
  g1[5] = K;                    // tensor_dim0_stride[31:0]  (bits 191:160)
  g1[6] = 0;
  g1[7] = 0;
  i32x4 z4 = { 0, 0, 0, 0 };
  i32x8 z8 = { 0, 0, 0, 0, 0, 0, 0, 0 };
  __builtin_amdgcn_tensor_load_to_lds(g0, g1, z4, z4, z8, 0);
}
#endif

// ---------------------------------------------------------------------------
// f32 -> bf16 elementwise convert (for weights)
// ---------------------------------------------------------------------------
__global__ void cvt_f32_bf16(const float* __restrict__ a, bf16_t* __restrict__ o,
                             long long n)
{
  long long i = (long long)blockIdx.x * blockDim.x + threadIdx.x;
  long long stride = (long long)gridDim.x * blockDim.x;
  for (; i < n; i += stride) o[i] = (bf16_t)a[i];
}

// ---------------------------------------------------------------------------
// LayerNorm: one block per row, f32 in -> bf16 out
// ---------------------------------------------------------------------------
__global__ __launch_bounds__(256)
void ln_kernel(const float* __restrict__ x, const float* __restrict__ wgt,
               const float* __restrict__ bia, bf16_t* __restrict__ out, int C)
{
  const int row = blockIdx.x;
  const int tid = threadIdx.x;
  const float* xr = x + (size_t)row * C;
  float s = 0.f, s2 = 0.f;
  for (int i = tid; i < C; i += 256) { float v = xr[i]; s += v; s2 += v * v; }
#pragma unroll
  for (int off = 16; off >= 1; off >>= 1) {
    s  += __shfl_xor(s,  off, 32);
    s2 += __shfl_xor(s2, off, 32);
  }
  __shared__ float ps[8], ps2[8];
  if ((tid & 31) == 0) { ps[tid >> 5] = s; ps2[tid >> 5] = s2; }
  __syncthreads();
  s = 0.f; s2 = 0.f;
#pragma unroll
  for (int i = 0; i < 8; i++) { s += ps[i]; s2 += ps2[i]; }
  const float mean = s / C;
  const float var  = s2 / C - mean * mean;
  const float rstd = rsqrtf(var + 1e-5f);
  for (int i = tid; i < C; i += 256)
    out[(size_t)row * C + i] = (bf16_t)((xr[i] - mean) * rstd * wgt[i] + bia[i]);
}

// ---------------------------------------------------------------------------
// WMMA bf16 GEMM:  out[M,N] = A[M,K] * W[N,K]^T + bias  (+ epilogue)
// Block tile 128x128, 8 waves (2x4), wave tile 64x32 = 4x2 WMMA tiles.
// Double-buffered LDS; tiles staged by TDM (or per-thread loads as fallback).
// LDS rows padded to 40 bf16 (20 dwords): 16 rows hit 16 distinct bank sets.
// ---------------------------------------------------------------------------
enum { EPI_QKV = 0, EPI_RESID = 1, EPI_GAUSS = 2 };

template <int MODE>
__global__ __launch_bounds__(256)
void gemm_bf16(const bf16_t* __restrict__ A, const bf16_t* __restrict__ W,
               const float* __restrict__ bias, const float* __restrict__ resid,
               bf16_t* __restrict__ outB, float* __restrict__ outF,
               int M, int N, int K,
               const float* __restrict__ muP, const float* __restrict__ sigP,
               const float* __restrict__ gamP, const float* __restrict__ betP)
{
  __shared__ bf16_t sA[2][128 * 40];
  __shared__ bf16_t sW[2][128 * 40];
  const int tid  = threadIdx.x;
  const int lane = tid & 31, wv = tid >> 5;
  const int half = lane >> 4, lr = lane & 15;
  const int wm = (wv >> 2) * 64, wn = (wv & 3) * 32;
  const int bm = blockIdx.y * 128, bn = blockIdx.x * 128;

  // fallback cooperative loader coords (one uint4 = 8 bf16 per thread)
  const int lrow = tid >> 1, lseg = tid & 1;
  const size_t aBase = (size_t)(bm + lrow) * K + lseg * 16;
  const size_t wBase = (size_t)(bn + lrow) * K + lseg * 16;
  (void)aBase; (void)wBase;

#if USE_TDM
#define LOAD_TILE(kt, buf)                                                     \
  do {                                                                         \
    if (wv == 0) {                                                             \
      tdm_load_tile(A + (size_t)bm * K + (size_t)(kt) * 32,                    \
                    (unsigned)(uintptr_t)&sA[buf][0], K);                      \
      tdm_load_tile(W + (size_t)bn * K + (size_t)(kt) * 32,                    \
                    (unsigned)(uintptr_t)&sW[buf][0], K);                      \
    }                                                                          \
  } while (0)
#else
#define LOAD_TILE(kt, buf)                                                     \
  do {                                                                         \
    *(uint4*)(&sA[buf][lrow * 40 + lseg * 16]) =                               \
        *(const uint4*)(A + aBase + (size_t)(kt) * 32);                        \
    *(uint4*)(&sW[buf][lrow * 40 + lseg * 16]) =                               \
        *(const uint4*)(W + wBase + (size_t)(kt) * 32);                        \
    if ((kt) + 1 < nk) {                                                       \
      __builtin_prefetch(A + aBase + (size_t)((kt) + 1) * 32, 0, 1);           \
      __builtin_prefetch(W + wBase + (size_t)((kt) + 1) * 32, 0, 1);           \
    }                                                                          \
  } while (0)
#endif

  v8f acc[4][2] = {};
  const int nk = K >> 5;

  LOAD_TILE(0, 0);
  for (int kt = 0; kt < nk; ++kt) {
    const int cur = kt & 1;
    if (kt + 1 < nk) LOAD_TILE(kt + 1, cur ^ 1);   // overlap next DMA w/ compute
#if USE_TDM
    if (wv == 0) {   // wait for current tile's 2 DMAs (next tile's may pend)
      if (kt + 1 < nk) __builtin_amdgcn_s_wait_tensorcnt(2);
      else             __builtin_amdgcn_s_wait_tensorcnt(0);
    }
#endif
    __syncthreads();

    const unsigned int* sAu = (const unsigned int*)&sA[cur][0];
    const unsigned int* sWu = (const unsigned int*)&sW[cur][0];
    Frag16 af[4], wf[2];
#pragma unroll
    for (int i = 0; i < 4; i++) {
      const unsigned int* ap = sAu + (wm + i * 16 + lr) * 20;
#pragma unroll
      for (int v = 0; v < 4; v++) {      // A 16x32 bf16 ISA layout
        af[i].u[v]     = ap[half * 4 + v];       // K 0-7 / 8-15
        af[i].u[4 + v] = ap[8 + half * 4 + v];   // K 16-23 / 24-31
      }
    }
#pragma unroll
    for (int j = 0; j < 2; j++) {
      const unsigned int* wp = sWu + (wn + j * 16 + lr) * 20;
#pragma unroll
      for (int v = 0; v < 8; v++) wf[j].u[v] = wp[(v >> 2) * 8 + half * 4 + (v & 3)];  // B 32x16
    }
#pragma unroll
    for (int i = 0; i < 4; i++)
#pragma unroll
      for (int j = 0; j < 2; j++)
        acc[i][j] = WMMA_BF16(af[i].v, wf[j].v, acc[i][j]);
    __syncthreads();
  }
#undef LOAD_TILE

  float mu = 0.f, sg = 1.f, ga = 1.f, be = 0.f;
  if (MODE == EPI_GAUSS) {
    mu = *muP; sg = fabsf(*sigP) + 1e-8f; ga = *gamP; be = *betP;
  }
#pragma unroll
  for (int i = 0; i < 4; i++) {
#pragma unroll
    for (int j = 0; j < 2; j++) {
      const int col = bn + wn + j * 16 + lr;
      const float bcol = bias[col];
#pragma unroll
      for (int r = 0; r < 8; r++) {
        const int row = bm + wm + i * 16 + half * 8 + r;  // C/D layout
        float v = acc[i][j][r] + bcol;
        const size_t idx = (size_t)row * N + col;
        if (MODE == EPI_QKV) {
          outB[idx] = (bf16_t)v;
        } else if (MODE == EPI_RESID) {
          outF[idx] = v + resid[idx];
        } else {
          float t = v - mu;
          outB[idx] = (bf16_t)(ga * expf(-(t * t) / (2.f * sg * sg)) + be);
        }
      }
    }
  }
}

// ---------------------------------------------------------------------------
// Flash attention: block = (64-query tile, head, batch), 4 waves x 16 queries
// ---------------------------------------------------------------------------
__global__ __launch_bounds__(128)
void attn_kernel(const bf16_t* __restrict__ qkv, bf16_t* __restrict__ y)
{
  constexpr int T = 2048, C = 1024, HS = 64;
  __shared__ bf16_t sK[64 * 72];      // [key][hs], row pad 72 bf16 (36 dw)
  __shared__ bf16_t sV[64 * 72];      // transposed: [hs][key]
  __shared__ bf16_t sP[4][16 * 40];   // per-wave P tile [query][key]
  const int tid  = threadIdx.x;
  const int lane = tid & 31, w = tid >> 5;
  const int half = lane >> 4, lr = lane & 15;
  const int qt = blockIdx.x, h = blockIdx.y, b = blockIdx.z;
  const int q0 = qt * 64 + w * 16;
  const float scale = 0.125f;  // 1/sqrt(64)

  // Q fragments (loop-invariant), direct from global
  Frag16 qf[2];
  {
    const unsigned int* qp =
        (const unsigned int*)(qkv + ((size_t)(b * T + q0 + lr)) * (3 * C) + h * HS);
#pragma unroll
    for (int v = 0; v < 4; v++) {
      qf[0].u[v]     = qp[half * 4 + v];
      qf[0].u[4 + v] = qp[8 + half * 4 + v];
      qf[1].u[v]     = qp[16 + half * 4 + v];
      qf[1].u[4 + v] = qp[24 + half * 4 + v];
    }
  }
  float mrow[8], lsum[8];
#pragma unroll
  for (int r = 0; r < 8; r++) { mrow[r] = -1e30f; lsum[r] = 0.f; }
  v8f oacc[4] = {};
  const unsigned int* sKu = (const unsigned int*)sK;
  const unsigned int* sVu = (const unsigned int*)sV;

  for (int kt = 0; kt <= qt; ++kt) {
    __syncthreads();
    // K tile (row-major == B-operand source)
#pragma unroll
    for (int it = 0; it < 4; ++it) {
      int idx = it * 128 + tid;
      int row = idx >> 3, seg = idx & 7;
      *(uint4*)(sK + row * 72 + seg * 8) =
          *(const uint4*)(qkv + ((size_t)(b * T + kt * 64 + row)) * (3 * C) +
                          C + h * HS + seg * 8);
    }
    // V tile, transposed into LDS so PV B-operand reads are contiguous
#pragma unroll
    for (int it = 0; it < 16; ++it) {
      int idx = it * 128 + tid;
      int row = idx >> 5;
      int dp  = (idx & 31) * 2;
      CvtU cv;
      cv.u = *(const unsigned int*)(qkv + ((size_t)(b * T + kt * 64 + row)) * (3 * C) +
                                    2 * C + h * HS + dp);
      sV[(dp    ) * 72 + row] = cv.h[0];
      sV[(dp + 1) * 72 + row] = cv.h[1];
    }
    __syncthreads();

    // S = Q K^T  (M=16 queries, N=64 keys, K=hs=64 -> 2 wmma k-steps)
    v8f sacc[4] = {};
#pragma unroll
    for (int c = 0; c < 2; c++) {
#pragma unroll
      for (int j = 0; j < 4; j++) {
        Frag16 kf;
        const unsigned int* kp = sKu + (j * 16 + lr) * 36;
#pragma unroll
        for (int v = 0; v < 8; v++) kf.u[v] = kp[16 * c + (v >> 2) * 8 + half * 4 + (v & 3)];
        sacc[j] = WMMA_BF16(qf[c].v, kf.v, sacc[j]);
      }
    }

    // streaming causal softmax
#pragma unroll
    for (int r = 0; r < 8; r++) {
      const int qr = q0 + half * 8 + r;
      float rmax = -1e30f;
#pragma unroll
      for (int j = 0; j < 4; j++) {
        const int key = kt * 64 + j * 16 + lr;
        float lg = sacc[j][r] * scale;
        if (key > qr) lg = -1e30f;
        sacc[j][r] = lg;
        rmax = fmaxf(rmax, lg);
      }
#pragma unroll
      for (int off = 8; off >= 1; off >>= 1)
        rmax = fmaxf(rmax, __shfl_xor(rmax, off, 32));
      const float mnew  = fmaxf(mrow[r], rmax);
      const float alpha = expf(mrow[r] - mnew);
      float rs = 0.f;
#pragma unroll
      for (int j = 0; j < 4; j++) {
        float p = expf(sacc[j][r] - mnew);
        sacc[j][r] = p;
        rs += p;
      }
#pragma unroll
      for (int off = 8; off >= 1; off >>= 1)
        rs += __shfl_xor(rs, off, 32);
      lsum[r] = lsum[r] * alpha + rs;
      mrow[r] = mnew;
#pragma unroll
      for (int j = 0; j < 4; j++) oacc[j][r] *= alpha;
      // C-layout -> A-layout via LDS bounce
#pragma unroll
      for (int j = 0; j < 4; j++)
        sP[w][(half * 8 + r) * 40 + j * 16 + lr] = (bf16_t)sacc[j][r];
    }
    __syncthreads();

    // O += P V  (K-dim = 64 keys -> 2 wmma k-steps)
    const unsigned int* pp = (const unsigned int*)(&sP[w][0]) + lr * 20;
#pragma unroll
    for (int c = 0; c < 2; c++) {
      Frag16 pf;
#pragma unroll
      for (int v = 0; v < 4; v++) {
        pf.u[v]     = pp[16 * c + half * 4 + v];
        pf.u[4 + v] = pp[16 * c + 8 + half * 4 + v];
      }
#pragma unroll
      for (int j = 0; j < 4; j++) {
        Frag16 vf;
        const unsigned int* vp = sVu + (j * 16 + lr) * 36;
#pragma unroll
        for (int v = 0; v < 8; v++) vf.u[v] = vp[16 * c + (v >> 2) * 8 + half * 4 + (v & 3)];
        oacc[j] = WMMA_BF16(pf.v, vf.v, oacc[j]);
      }
    }
  }

  // y = O / l
#pragma unroll
  for (int j = 0; j < 4; j++) {
#pragma unroll
    for (int r = 0; r < 8; r++) {
      const int t = q0 + half * 8 + r;
      const int d = j * 16 + lr;
      y[((size_t)(b * T + t)) * C + h * HS + d] = (bf16_t)(oacc[j][r] / lsum[r]);
    }
  }
}

// ---------------------------------------------------------------------------
// host orchestration
// ---------------------------------------------------------------------------
extern "C" void kernel_launch(void* const* d_in, const int* in_sizes, int n_in,
                              void* d_out, int out_size, void* d_ws, size_t ws_size,
                              hipStream_t stream)
{
  const float* x      = (const float*)d_in[0];
  const float* ln1_w  = (const float*)d_in[1];
  const float* ln1_b  = (const float*)d_in[2];
  const float* attn_w = (const float*)d_in[3];
  const float* attn_b = (const float*)d_in[4];
  const float* proj_w = (const float*)d_in[5];
  const float* proj_b = (const float*)d_in[6];
  const float* ln2_w  = (const float*)d_in[7];
  const float* ln2_b  = (const float*)d_in[8];
  const float* fc_w   = (const float*)d_in[9];
  const float* fc_b   = (const float*)d_in[10];
  const float* fc2_w  = (const float*)d_in[11];
  const float* fc2_b  = (const float*)d_in[12];
  const float* muP    = (const float*)d_in[13];
  const float* sigP   = (const float*)d_in[14];
  const float* gamP   = (const float*)d_in[15];
  const float* betP   = (const float*)d_in[16];
  (void)in_sizes; (void)n_in; (void)out_size; (void)ws_size;

  constexpr int B = 2, T = 2048, C = 1024, H = 4096, NH = 16;
  const int M = B * T;

  char* ws = (char*)d_ws;
  size_t off = 0;
  auto alloc = [&](size_t bytes) {
    char* p = ws + off;
    off += (bytes + 255) & ~(size_t)255;
    return p;
  };
  bf16_t* attn_wb = (bf16_t*)alloc((size_t)3 * C * C * 2);
  bf16_t* proj_wb = (bf16_t*)alloc((size_t)C * C * 2);
  bf16_t* fc_wb   = (bf16_t*)alloc((size_t)H * C * 2);
  bf16_t* fc2_wb  = (bf16_t*)alloc((size_t)C * H * 2);
  bf16_t* a_ln1   = (bf16_t*)alloc((size_t)M * C * 2);
  bf16_t* qkvB    = (bf16_t*)alloc((size_t)M * 3 * C * 2);
  bf16_t* yB      = (bf16_t*)alloc((size_t)M * C * 2);
  float*  x2      = (float*)alloc((size_t)M * C * 4);
  bf16_t* a_ln2   = (bf16_t*)alloc((size_t)M * C * 2);
  bf16_t* hact    = (bf16_t*)alloc((size_t)M * H * 2);

  // weight conversions
  cvt_f32_bf16<<<1024, 256, 0, stream>>>(attn_w, attn_wb, (long long)3 * C * C);
  cvt_f32_bf16<<<1024, 256, 0, stream>>>(proj_w, proj_wb, (long long)C * C);
  cvt_f32_bf16<<<1024, 256, 0, stream>>>(fc_w,   fc_wb,   (long long)H * C);
  cvt_f32_bf16<<<1024, 256, 0, stream>>>(fc2_w,  fc2_wb,  (long long)C * H);

  // LN1
  ln_kernel<<<M, 256, 0, stream>>>(x, ln1_w, ln1_b, a_ln1, C);

  // QKV GEMM: [M,C] x [3C,C]^T -> bf16
  gemm_bf16<EPI_QKV><<<dim3(3 * C / 128, M / 128), 256, 0, stream>>>(
      a_ln1, attn_wb, attn_b, nullptr, qkvB, nullptr, M, 3 * C, C,
      nullptr, nullptr, nullptr, nullptr);

  // attention
  attn_kernel<<<dim3(T / 64, NH, B), 128, 0, stream>>>(qkvB, yB);

  // proj GEMM + residual -> x2 (f32)
  gemm_bf16<EPI_RESID><<<dim3(C / 128, M / 128), 256, 0, stream>>>(
      yB, proj_wb, proj_b, x, nullptr, x2, M, C, C,
      nullptr, nullptr, nullptr, nullptr);

  // LN2
  ln_kernel<<<M, 256, 0, stream>>>(x2, ln2_w, ln2_b, a_ln2, C);

  // fc GEMM + gaussian activation -> bf16
  gemm_bf16<EPI_GAUSS><<<dim3(H / 128, M / 128), 256, 0, stream>>>(
      a_ln2, fc_wb, fc_b, nullptr, hact, nullptr, M, H, C,
      muP, sigP, gamP, betP);

  // fc2 GEMM + residual -> d_out (f32)
  gemm_bf16<EPI_RESID><<<dim3(C / 128, M / 128), 256, 0, stream>>>(
      hact, fc2_wb, fc2_b, x2, nullptr, (float*)d_out, M, C, H,
      nullptr, nullptr, nullptr, nullptr);
}
